// LDABaseline_7851200217279
// MI455X (gfx1250) — matrix operand order repaired and checked by reading, not compiled, not run on hardware
//
#include <hip/hip_runtime.h>
#include <hip/hip_bf16.h>

// ---------------------------------------------------------------------------
// Deformable local attention for MI455X (gfx1250, wave32, WMMA).
// B=4, NH=8, DH=32, G=4, C=256, CG=64, H=W=64, K=9, KS=5.
// GEMMs (q/k/v proj + out proj) via v_wmma_f32_16x16x32_f16 with the shared
// X panel staged into LDS through GLOBAL_LOAD_ASYNC_TO_LDS_B128 (ASYNCcnt).
// ---------------------------------------------------------------------------

typedef __attribute__((ext_vector_type(16))) _Float16 v16h;
typedef __attribute__((ext_vector_type(8)))  _Float16 h8;
typedef __attribute__((ext_vector_type(8)))  float    v8f;

#define NB   4
#define CC   256
#define HH   64
#define WW   64
#define HWSZ 4096
#define NHD  8
#define DHD  32
#define GG   4
#define CGC  64
#define KK   9

// ---- fragment loaders (k = 16*half + e in both A and B) -----------

// A: 16x32 f16, row-major weight tile.  Lane (half, m): row m, two runs of 8:
//   elements 0..15 -> K = 16*half + e
__device__ inline v16h load_a_frag(const _Float16* __restrict__ p, int half) {
  h8 r1 = *(const h8*)(p + 16 * half);
  h8 r2 = *(const h8*)(p + 16 * half + 8);
  v16h a;
#pragma unroll
  for (int i = 0; i < 8; ++i) { a[i] = r1[i]; a[8 + i] = r2[i]; }
  return a;
}

// B: 32x16 f16 (row n contiguous in K). Lane (half, n):
//   elements 0..15 -> K = 16*half + e   (one contiguous 32B read, from LDS)
__device__ inline v16h load_b_frag_lds(const _Float16* p, int half) {
  return *(const v16h*)(p + 16 * half);
}

// Async 16B global->LDS copy (GLOBAL_LOAD_ASYNC_TO_LDS_B128, ASYNCcnt).
__device__ inline void async_copy_b128(unsigned lds_off, const void* gaddr) {
  asm volatile("global_load_async_to_lds_b128 %0, %1, off"
               :: "v"(lds_off), "v"(gaddr) : "memory");
}
__device__ inline void wait_asynccnt0() {
  asm volatile("s_wait_asynccnt 0x0" ::: "memory");
}

// ---- conversion kernels ---------------------------------------------------

__global__ __launch_bounds__(256) void cvt_w_kernel(
    const float* __restrict__ wq, const float* __restrict__ wk,
    const float* __restrict__ wv, const float* __restrict__ wo,
    _Float16* __restrict__ wqh, _Float16* __restrict__ wkh,
    _Float16* __restrict__ wvh, _Float16* __restrict__ woh) {
  int i = blockIdx.x * 256 + threadIdx.x;
  if (i >= CC * CC) return;
  wqh[i] = (_Float16)wq[i];
  wkh[i] = (_Float16)wk[i];
  wvh[i] = (_Float16)wv[i];
  woh[i] = (_Float16)wo[i];
}

// x[B,C,HW] f32 -> xt[B,HW,C] f16 (transpose so B-fragments are contiguous)
__global__ __launch_bounds__(256) void cvt_x_kernel(
    const float* __restrict__ x, _Float16* __restrict__ xt) {
  int i = blockIdx.x * 256 + threadIdx.x;           // [0, B*HW*C)
  if (i >= NB * HWSZ * CC) return;
  int c  = i & (CC - 1);
  int r  = i >> 8;
  int hw = r & (HWSZ - 1);
  int b  = r >> 12;
  xt[i] = (_Float16)x[((size_t)(b * CC + c)) * HWSZ + hw];
}

// ---- fused QKV projection: 3 GEMMs sharing the LDS-staged X panel ---------

__global__ __launch_bounds__(128) void qkv_gemm_kernel(
    const _Float16* __restrict__ xt,
    const _Float16* __restrict__ wqh, const _Float16* __restrict__ wkh,
    const _Float16* __restrict__ wvh,
    const float* __restrict__ bq, const float* __restrict__ bk,
    const float* __restrict__ bv,
    float* __restrict__ qo, float* __restrict__ ko, float* __restrict__ vo) {
  __shared__ alignas(32) _Float16 xs[16][CC];   // 8 KB: 16 columns x K=256
  const int tid  = threadIdx.x;
  const int lane = tid & 31;
  const int wave = tid >> 5;
  const int half = lane >> 4;
  const int l16  = lane & 15;
  const int mo = blockIdx.y * 4 + wave;   // 0..15  (M tile of 16 out-channels)
  const int no = blockIdx.x;              // 0..255 (N tile of 16 columns)
  const int b  = blockIdx.z;

  // ---- stage the 16x256 f16 X panel into LDS with async copies ----
  {
    const _Float16* gbase = xt + ((size_t)(b * HWSZ + no * 16)) * CC;
#pragma unroll
    for (int i = 0; i < 4; ++i) {
      int chunk = tid + i * 128;            // 512 chunks of 16B
      int row = chunk >> 5;                 // column index 0..15
      int seg = chunk & 31;                 // 8-halfword segment
      async_copy_b128((unsigned)(size_t)(&xs[row][seg * 8]),
                      gbase + (size_t)row * CC + seg * 8);
    }
    wait_asynccnt0();
    __syncthreads();
  }

  const int arow = mo * 16 + l16;
  const int col  = no * 16 + l16;
  const _Float16* __restrict__ aq = wqh + (size_t)arow * CC;
  const _Float16* __restrict__ ak = wkh + (size_t)arow * CC;
  const _Float16* __restrict__ av = wvh + (size_t)arow * CC;
  const _Float16* brow = &xs[l16][0];

  v8f accq = {}; v8f acck = {}; v8f accv = {};

#pragma unroll
  for (int k0 = 0; k0 < CC; k0 += 32) {
    if (k0 + 32 < CC) {
      __builtin_prefetch(aq + k0 + 32, 0, 0);
      __builtin_prefetch(ak + k0 + 32, 0, 0);
      __builtin_prefetch(av + k0 + 32, 0, 0);
    }
    v16h bf = load_b_frag_lds(brow + k0, half);
    v16h fa = load_a_frag(aq + k0, half);
    accq = __builtin_amdgcn_wmma_f32_16x16x32_f16(false, fa, false, bf,
                                                  (short)0, accq, false, false);
    fa = load_a_frag(ak + k0, half);
    acck = __builtin_amdgcn_wmma_f32_16x16x32_f16(false, fa, false, bf,
                                                  (short)0, acck, false, false);
    fa = load_a_frag(av + k0, half);
    accv = __builtin_amdgcn_wmma_f32_16x16x32_f16(false, fa, false, bf,
                                                  (short)0, accv, false, false);
  }

#pragma unroll
  for (int r = 0; r < 8; ++r) {
    int m = mo * 16 + r + 8 * half;              // C/D: VGPR r -> M = r+8*half
    size_t idx = ((size_t)(b * CC + m)) * HWSZ + col;
    qo[idx] = accq[r] + bq[m];
    ko[idx] = acck[r] + bk[m];
    vo[idx] = accv[r] + bv[m];
  }
}

// ---- output projection GEMM ----------------------------------------------

__global__ __launch_bounds__(128) void out_gemm_kernel(
    const _Float16* __restrict__ att, const _Float16* __restrict__ woh,
    const float* __restrict__ bo, float* __restrict__ yo) {
  __shared__ alignas(32) _Float16 xs[16][CC];
  const int tid  = threadIdx.x;
  const int lane = tid & 31;
  const int wave = tid >> 5;
  const int half = lane >> 4;
  const int l16  = lane & 15;
  const int mo = blockIdx.y * 4 + wave;
  const int no = blockIdx.x;
  const int b  = blockIdx.z;

  {
    const _Float16* gbase = att + ((size_t)(b * HWSZ + no * 16)) * CC;
#pragma unroll
    for (int i = 0; i < 4; ++i) {
      int chunk = tid + i * 128;
      int row = chunk >> 5;
      int seg = chunk & 31;
      async_copy_b128((unsigned)(size_t)(&xs[row][seg * 8]),
                      gbase + (size_t)row * CC + seg * 8);
    }
    wait_asynccnt0();
    __syncthreads();
  }

  const int arow = mo * 16 + l16;
  const int col  = no * 16 + l16;
  const _Float16* __restrict__ aw = woh + (size_t)arow * CC;
  const _Float16* brow = &xs[l16][0];

  v8f acc = {};
#pragma unroll
  for (int k0 = 0; k0 < CC; k0 += 32) {
    if (k0 + 32 < CC) __builtin_prefetch(aw + k0 + 32, 0, 0);
    v16h bf = load_b_frag_lds(brow + k0, half);
    v16h fa = load_a_frag(aw + k0, half);
    acc = __builtin_amdgcn_wmma_f32_16x16x32_f16(false, fa, false, bf,
                                                 (short)0, acc, false, false);
  }
#pragma unroll
  for (int r = 0; r < 8; ++r) {
    int m = mo * 16 + r + 8 * half;
    yo[((size_t)(b * CC + m)) * HWSZ + col] = acc[r] + bo[m];
  }
}

// ---- reference sampling points (output #3) --------------------------------

__global__ __launch_bounds__(256) void ref_kernel(float* __restrict__ refo) {
  int idx = blockIdx.x * 256 + threadIdx.x;
  if (idx >= 2 * KK * HWSZ) return;
  int pix = idx & (HWSZ - 1);
  int ch  = idx >> 12;
  int coord = ch / KK, kk = ch % KK, fi = kk / 3, fj = kk % 3;
  int y = pix >> 6, x = pix & 63;
  int t = (coord == 0) ? min(max(y + fi - 1, 0), HH - 1)
                       : min(max(x + fj - 1, 0), WW - 1);
  refo[idx] = (float)t * (2.0f / 63.0f) - 1.0f;
}

// ---- depthwise5x5 + LN + GELU + pointwise offsets -> pos (output #2) ------

__global__ __launch_bounds__(128) void offset_kernel(
    const float* __restrict__ q, const float* __restrict__ dw_w,
    const float* __restrict__ dw_b, const float* __restrict__ ln_w,
    const float* __restrict__ ln_b, const float* __restrict__ pw_w,
    float* __restrict__ pos) {
  __shared__ float sh[4][CGC];
  const int wave = threadIdx.x >> 5, lane = threadIdx.x & 31;
  const int wgid = blockIdx.x * 4 + wave;          // [0, 16*4096)
  const int bg = wgid >> 12;
  const int pix = wgid & (HWSZ - 1);
  const int y = pix >> 6, x = pix & 63;

  float hv[2];
#pragma unroll
  for (int t = 0; t < 2; ++t) {
    int c = lane + t * 32;
    const float* __restrict__ qc  = q + ((size_t)(bg * CGC + c)) * HWSZ;
    const float* __restrict__ wcc = dw_w + c * 25;
    float acc = dw_b[c];
#pragma unroll
    for (int dy = 0; dy < 5; ++dy) {
      int yy = y + dy - 2;
      if (yy < 0 || yy >= HH) continue;
#pragma unroll
      for (int dx = 0; dx < 5; ++dx) {
        int xx = x + dx - 2;
        if (xx < 0 || xx >= WW) continue;
        acc += qc[yy * WW + xx] * wcc[dy * 5 + dx];
      }
    }
    hv[t] = acc;
  }
  // wave32 butterfly reduction for mean / var over 64 channels
  float s = hv[0] + hv[1], ss = hv[0] * hv[0] + hv[1] * hv[1];
#pragma unroll
  for (int off = 16; off > 0; off >>= 1) {
    s  += __shfl_xor(s, off, 32);
    ss += __shfl_xor(ss, off, 32);
  }
  float mu   = s * (1.0f / 64.0f);
  float var  = ss * (1.0f / 64.0f) - mu * mu;
  float rinv = rsqrtf(var + 1e-5f);
#pragma unroll
  for (int t = 0; t < 2; ++t) {
    int c = lane + t * 32;
    float hn = (hv[t] - mu) * rinv * ln_w[c] + ln_b[c];
    sh[wave][c] = 0.5f * hn * (1.0f + erff(hn * 0.70710678118654752f)); // exact gelu
  }
  __syncthreads();
  if (lane < 2 * KK) {
    const int o = lane;
    const float* __restrict__ pw = pw_w + o * CGC;
    float offv = 0.0f;
#pragma unroll 16
    for (int c = 0; c < CGC; ++c) offv += pw[c] * sh[wave][c];
    int coord = o / KK, kk = o % KK, fi = kk / 3, fj = kk % 3;
    int t = (coord == 0) ? min(max(y + fi - 1, 0), HH - 1)
                         : min(max(x + fj - 1, 0), WW - 1);
    float refv = (float)t * (2.0f / 63.0f) - 1.0f;
    float pv = fminf(fmaxf(offv + refv, -1.0f), 1.0f);
    pos[(((size_t)bg * HWSZ + pix) * KK + kk) * 2 + coord] = pv;
  }
}

// ---- deformable attention: sample k/v, dot, rpe bias, softmax, gather -----

__global__ __launch_bounds__(128) void attn_kernel(
    const float* __restrict__ q, const float* __restrict__ kbuf,
    const float* __restrict__ vbuf, const float* __restrict__ pos,
    const float* __restrict__ rpe, _Float16* __restrict__ att) {
  const int wgid = blockIdx.x * 4 + (threadIdx.x >> 5);  // [0, B*NH*HW)
  const int lane = threadIdx.x & 31;                     // d channel
  const int qpos = wgid & (HWSZ - 1);
  const int bh   = wgid >> 12;
  const int h = bh & 7, b = bh >> 3;
  const int bg = b * GG + (h >> 1);

  const float SCALE = 0.17677669529663687f;  // 32^-0.5
  const size_t cbase = ((size_t)(b * CC + h * DHD + lane)) * HWSZ;
  const float qv = q[cbase + qpos] * SCALE;
  const int yq = qpos >> 6, xq = qpos & 63;
  const float qgy = (float)yq * (2.0f / 63.0f) - 1.0f;
  const float qgx = (float)xq * (2.0f / 63.0f) - 1.0f;
  const float* __restrict__ posp = pos + ((size_t)(bg * HWSZ + qpos)) * (KK * 2);
  const float* __restrict__ rh   = rpe + (size_t)h * 127 * 127;
  const float* __restrict__ kb   = kbuf + cbase;
  const float* __restrict__ vb   = vbuf + cbase;

  float logit[KK], vs[KK];
#pragma unroll
  for (int k = 0; k < KK; ++k) {
    float py = posp[k * 2 + 0];
    float px = posp[k * 2 + 1];
    // bilinear sample of k/v on 64x64 (grid takes x from pos[...,1])
    float xs = (px + 1.0f) * 0.5f * 63.0f;
    float ys = (py + 1.0f) * 0.5f * 63.0f;
    float x0f = floorf(xs), y0f = floorf(ys);
    float wx = xs - x0f, wy = ys - y0f;
    int x0 = min(max((int)x0f, 0), 63), x1 = min(x0 + 1, 63);
    int y0 = min(max((int)y0f, 0), 63), y1 = min(y0 + 1, 63);
    float w00 = (1.0f - wx) * (1.0f - wy), w01 = wx * (1.0f - wy);
    float w10 = (1.0f - wx) * wy,          w11 = wx * wy;
    float ksmp = kb[y0 * WW + x0] * w00 + kb[y0 * WW + x1] * w01 +
                 kb[y1 * WW + x0] * w10 + kb[y1 * WW + x1] * w11;
    vs[k]      = vb[y0 * WW + x0] * w00 + vb[y0 * WW + x1] * w01 +
                 vb[y1 * WW + x0] * w10 + vb[y1 * WW + x1] * w11;
    // q . k over DH=32 via wave32 butterfly (result in every lane)
    float part = qv * ksmp;
#pragma unroll
    for (int off = 16; off > 0; off >>= 1) part += __shfl_xor(part, off, 32);
    // rpe bias: sample rpe[h] (127x127) at disp (x from [...,1] after reverse)
    float dyv = (qgy - py) * 0.5f, dxv = (qgx - px) * 0.5f;
    float xr = (dxv + 1.0f) * 0.5f * 126.0f;
    float yr = (dyv + 1.0f) * 0.5f * 126.0f;
    float xr0f = floorf(xr), yr0f = floorf(yr);
    float bwx = xr - xr0f, bwy = yr - yr0f;
    int bx0 = min(max((int)xr0f, 0), 126), bx1 = min(bx0 + 1, 126);
    int by0 = min(max((int)yr0f, 0), 126), by1 = min(by0 + 1, 126);
    float bias = rh[by0 * 127 + bx0] * (1.0f - bwx) * (1.0f - bwy) +
                 rh[by0 * 127 + bx1] * bwx * (1.0f - bwy) +
                 rh[by1 * 127 + bx0] * (1.0f - bwx) * bwy +
                 rh[by1 * 127 + bx1] * bwx * bwy;
    logit[k] = part + bias;
  }
  // softmax over K=9 (redundant per lane)
  float mx = logit[0];
#pragma unroll
  for (int k = 1; k < KK; ++k) mx = fmaxf(mx, logit[k]);
  float sum = 0.0f;
#pragma unroll
  for (int k = 0; k < KK; ++k) { logit[k] = __expf(logit[k] - mx); sum += logit[k]; }
  float o = 0.0f;
#pragma unroll
  for (int k = 0; k < KK; ++k) o += logit[k] * vs[k];
  o /= sum;
  // store transposed f16 [B,HW,C] so out_gemm B-fragments are contiguous
  att[((size_t)(b * HWSZ + qpos)) * CC + h * DHD + lane] = (_Float16)o;
}

// ---------------------------------------------------------------------------

extern "C" void kernel_launch(void* const* d_in, const int* in_sizes, int n_in,
                              void* d_out, int out_size, void* d_ws, size_t ws_size,
                              hipStream_t stream) {
  const float* x    = (const float*)d_in[0];
  const float* wq   = (const float*)d_in[1];
  const float* bq   = (const float*)d_in[2];
  const float* wk   = (const float*)d_in[3];
  const float* bk   = (const float*)d_in[4];
  const float* wv   = (const float*)d_in[5];
  const float* bv   = (const float*)d_in[6];
  const float* wo   = (const float*)d_in[7];
  const float* bo   = (const float*)d_in[8];
  const float* dw_w = (const float*)d_in[9];
  const float* dw_b = (const float*)d_in[10];
  const float* ln_w = (const float*)d_in[11];
  const float* ln_b = (const float*)d_in[12];
  const float* pw_w = (const float*)d_in[13];
  const float* rpe  = (const float*)d_in[14];

  char* ws = (char*)d_ws;
  _Float16* wqh = (_Float16*)(ws + 0);
  _Float16* wkh = (_Float16*)(ws + 131072);
  _Float16* wvh = (_Float16*)(ws + 262144);
  _Float16* woh = (_Float16*)(ws + 393216);
  _Float16* xt  = (_Float16*)(ws + 524288);      //  8 MB
  float*    qb  = (float*)(ws + 8912896);        // 16 MB
  float*    kb  = (float*)(ws + 25690112);       // 16 MB
  float*    vb  = (float*)(ws + 42467328);       // 16 MB
  _Float16* att = (_Float16*)(ws + 59244544);    //  8 MB  (total ~64.5 MB)

  float* yout = (float*)d_out;                   // (B,C,H,W)  = 4194304 f32
  float* posO = yout + 4194304;                  // (BG,HW,K,2)= 1179648 f32
  float* refO = yout + 5373952;                  // (1,2K,H,W) =   73728 f32

  cvt_w_kernel<<<dim3(256), dim3(256), 0, stream>>>(wq, wk, wv, wo,
                                                    wqh, wkh, wvh, woh);
  cvt_x_kernel<<<dim3(16384), dim3(256), 0, stream>>>(x, xt);
  qkv_gemm_kernel<<<dim3(256, 4, 4), dim3(128), 0, stream>>>(
      xt, wqh, wkh, wvh, bq, bk, bv, qb, kb, vb);
  ref_kernel<<<dim3(288), dim3(256), 0, stream>>>(refO);
  offset_kernel<<<dim3(16384), dim3(128), 0, stream>>>(qb, dw_w, dw_b, ln_w,
                                                       ln_b, pw_w, posO);
  attn_kernel<<<dim3(32768), dim3(128), 0, stream>>>(qb, kb, vb, posO, rpe, att);
  out_gemm_kernel<<<dim3(256, 4, 4), dim3(128), 0, stream>>>(att, woh, bo, yout);
}
